// ExpDock_35347580846427
// MI455X (gfx1250) — matrix operand order repaired and checked by reading, not compiled, not run on hardware
//
#include <hip/hip_runtime.h>
#include <math.h>

namespace {

constexpr int cBS   = 4;
constexpr int cNR   = 2048;
constexpr int cNN   = 4096;
constexpr int cN    = 16384;      // BS * NN
constexpr int cKN   = 9;
constexpr int cE    = cN * cKN;   // 147456
constexpr int cH    = 128;
constexpr int cH2   = 256;
constexpr int cRBF  = 20;
constexpr int cL    = 4;
constexpr int cKP   = 128;
constexpr int cEIN  = 280;
constexpr int cEINP = 288;
constexpr float cSCALE = 0.015f;

typedef __bf16 bf16;
typedef __attribute__((ext_vector_type(16))) __bf16 v16bf;
typedef __attribute__((ext_vector_type(8)))  __bf16 v8bf;
typedef __attribute__((ext_vector_type(8)))  float  v8f;

__device__ __forceinline__ float siluf(float x){ return x / (1.f + __expf(-x)); }
__device__ __forceinline__ float sigmf(float x){ return 1.f / (1.f + __expf(-x)); }
__device__ __forceinline__ float huberf(float x){ float a = fabsf(x); return (a < 1.f) ? 0.5f*x*x : a - 0.5f; }

// ---------------- WMMA tile loaders (bf16, wave32) ----------------
// A 16x32 bf16 tile: lanes 0-15 -> rows, K {0..7,16..23}; lanes 16-31 -> K {8..15,24..31}
__device__ __forceinline__ v16bf load_a_tile(const bf16* __restrict__ base, int ld, int row, int kh){
  const bf16* p0 = base + (size_t)row * ld + kh * 8;
  v8bf lo = *(const v8bf*)p0;
  v8bf hi = *(const v8bf*)(p0 + 16);
  v16bf r;
#pragma unroll
  for (int i = 0; i < 8; i++){ r[i] = lo[i]; r[8+i] = hi[i]; }
  return r;
}
// B 32x16 tile from N-major (transposed-weight) storage: lane = column,
// lanes 0-15 hold K {0..7,16..23}, lanes 16-31 hold K {8..15,24..31}.
__device__ __forceinline__ v16bf load_b_tile(const bf16* __restrict__ base, int ld, int col, int kh){
  const bf16* p0 = base + (size_t)col * ld + kh * 8;
  v8bf lo = *(const v8bf*)p0;
  v8bf hi = *(const v8bf*)(p0 + 16);
  v16bf r;
#pragma unroll
  for (int i = 0; i < 8; i++){ r[i] = lo[i]; r[8+i] = hi[i]; }
  return r;
}

enum { EP_BIAS = 1, EP_SILU = 2, EP_RES = 4, EP_OUTF = 8, EP_OUTB = 16 };

// Generic GEMM: C[MxN] = epilogue(A[MxK] @ B[KxN]); BT is B^T stored [N x K] bf16.
// Block = 128 threads (4 waves); block tile 128x128; wave tile 64x64
// (4 A tiles x 4 B tiles -> 16 WMMAs per 32-wide K step; all 8 tile loads
// are issued before the WMMA burst so the matrix pipe is not per-tile fenced).
template<int F>
__global__ __launch_bounds__(128)
void k_gemm(const bf16* __restrict__ A, const bf16* __restrict__ BT,
            const float* __restrict__ bias, const float* __restrict__ resid,
            float* __restrict__ Cf, bf16* __restrict__ Cb,
            int M, int N, int K)
{
  const int lane   = threadIdx.x & 31;
  const int wave   = threadIdx.x >> 5;
  const int tilesN = N >> 7;
  const int bm = blockIdx.x / tilesN;
  const int bn = blockIdx.x - bm * tilesN;
  const int m0 = bm * 128 + (wave & 1) * 64;
  const int n0 = bn * 128 + (wave >> 1) * 64;
  const int r  = lane & 15;
  const int kh = lane >> 4;

  v8f acc[16];
#pragma unroll
  for (int t = 0; t < 16; t++)
#pragma unroll
    for (int i = 0; i < 8; i++) acc[t][i] = 0.f;

  const bf16* Abase = A + (size_t)m0 * K;
  const bf16* Bbase = BT + (size_t)n0 * K;
  for (int k0 = 0; k0 < K; k0 += 32){
    v16bf av[4], bv[4];
#pragma unroll
    for (int i = 0; i < 4; i++)
      av[i] = load_a_tile(Abase + (size_t)(i * 16) * K + k0, K, r, kh);
#pragma unroll
    for (int t = 0; t < 4; t++)
      bv[t] = load_b_tile(Bbase + (size_t)(t * 16) * K + k0, K, r, kh);
    __builtin_prefetch(Abase + (size_t)r * K + k0 + 32, 0, 1);
#pragma unroll
    for (int i = 0; i < 4; i++)
#pragma unroll
      for (int t = 0; t < 4; t++)
        acc[i * 4 + t] = __builtin_amdgcn_wmma_f32_16x16x32_bf16(
            false, av[i], false, bv[t], (short)0, acc[i * 4 + t], false, false);
  }

#pragma unroll
  for (int i = 0; i < 4; i++){
    const int rowb = m0 + i * 16 + kh * 8;
#pragma unroll
    for (int t = 0; t < 4; t++){
      const int col = n0 + t * 16 + r;
      const float bsv = (F & EP_BIAS) ? bias[col] : 0.f;
#pragma unroll
      for (int e = 0; e < 8; e++){
        const size_t idx = (size_t)(rowb + e) * N + col;
        float v = acc[i * 4 + t][e] + bsv;
        if (F & EP_SILU) v = siluf(v);
        if (F & EP_RES)  v += resid[idx];
        if (F & EP_OUTF) Cf[idx] = v;
        if (F & EP_OUTB) Cb[idx] = (bf16)v;
      }
    }
  }
}

// ---------------- small utility kernels ----------------
__global__ void k_zero(float* __restrict__ p, int n){
  int i = blockIdx.x * blockDim.x + threadIdx.x;
  if (i < n) p[i] = 0.f;
}
__global__ void k_cvt_bf16(const float* __restrict__ s, bf16* __restrict__ d, int n){
  int i = blockIdx.x * blockDim.x + threadIdx.x;
  if (i < n) d[i] = (bf16)s[i];
}
// W [K x N] f32  ->  WT [N x Kpad] bf16 (zero padded K)
__global__ void k_wT(const float* __restrict__ W, bf16* __restrict__ WT, int K, int N, int Kpad){
  int idx = blockIdx.x * blockDim.x + threadIdx.x;
  if (idx >= N * Kpad) return;
  int n = idx / Kpad, k = idx - n * Kpad;
  WT[idx] = (k < K) ? (bf16)W[(size_t)k * N + n] : (bf16)0.f;
}

__global__ void k_coords_nodes(const float* __restrict__ X, const int* __restrict__ Seg,
                               const int* __restrict__ bid, const float* __restrict__ center,
                               const float* __restrict__ rot, const float* __restrict__ trans,
                               const float* __restrict__ meanp, const float* __restrict__ stdp,
                               float* __restrict__ Xn, float* __restrict__ oXn){
  int i = blockIdx.x * blockDim.x + threadIdx.x;
  if (i >= cN) return;
  float sm = (stdp[0] + stdp[1] + stdp[2]) * (1.f / 3.f);
  int b = bid[i];
  float xc[3], ori[3];
#pragma unroll
  for (int d = 0; d < 3; d++){ xc[d] = X[i*3+d] - center[b*3+d]; ori[d] = xc[d]; }
  if (Seg[i] == 0){
    float xr[3];
#pragma unroll
    for (int e = 0; e < 3; e++)
      xr[e] = xc[0]*rot[b*9+e] + xc[1]*rot[b*9+3+e] + xc[2]*rot[b*9+6+e] + trans[b*3+e]*sm;
#pragma unroll
    for (int e = 0; e < 3; e++) xc[e] = xr[e];
  }
#pragma unroll
  for (int e = 0; e < 3; e++){
    Xn[i*3+e]  = (xc[e]  - meanp[e]) / stdp[e];
    oXn[i*3+e] = (ori[e] - meanp[e]) / stdp[e];
  }
}

__global__ void k_coords_kp(const float* __restrict__ keypoints, const int* __restrict__ k_bid,
                            const float* __restrict__ center, const float* __restrict__ rot,
                            const float* __restrict__ trans, const float* __restrict__ meanp,
                            const float* __restrict__ stdp,
                            float* __restrict__ kpn, float* __restrict__ tkpn){
  int i = blockIdx.x * blockDim.x + threadIdx.x;
  if (i >= cBS * cKP) return;
  float sm = (stdp[0] + stdp[1] + stdp[2]) * (1.f / 3.f);
  int b = k_bid[i];
  float kc[3], tk[3];
#pragma unroll
  for (int d = 0; d < 3; d++) kc[d] = keypoints[i*3+d] - center[b*3+d];
#pragma unroll
  for (int e = 0; e < 3; e++)
    tk[e] = kc[0]*rot[b*9+e] + kc[1]*rot[b*9+3+e] + kc[2]*rot[b*9+6+e] + trans[b*3+e]*sm;
#pragma unroll
  for (int e = 0; e < 3; e++){
    kpn[i*3+e]  = (kc[e] - meanp[e]) / stdp[e];
    tkpn[i*3+e] = (tk[e] - meanp[e]) / stdp[e];
  }
}

__global__ void k_knn(const float* __restrict__ Xn, int* __restrict__ src){
  int i = blockIdx.x * blockDim.x + threadIdx.x;
  if (i >= cN) return;
  int b = i >> 12, base = b << 12, li = i - base;
  float xi = Xn[i*3], yi = Xn[i*3+1], zi = Xn[i*3+2];
  float bd[cKN]; int bi2[cKN];
#pragma unroll
  for (int k = 0; k < cKN; k++){ bd[k] = 3.0e38f; bi2[k] = base; }
  const float* P = Xn + (size_t)base * 3;
  for (int j = 0; j < cNN; j++){
    float dx = P[j*3] - xi, dy = P[j*3+1] - yi, dz = P[j*3+2] - zi;
    float d2 = dx*dx + dy*dy + dz*dz;
    if (j == li) continue;
    if (d2 < bd[cKN-1]){
      int k = cKN - 1;
      while (k > 0 && bd[k-1] > d2){ bd[k] = bd[k-1]; bi2[k] = bi2[k-1]; k--; }
      bd[k] = d2; bi2[k] = base + j;
    }
  }
#pragma unroll
  for (int k = 0; k < cKN; k++) src[(size_t)i*cKN + k] = bi2[k];
}

__global__ void k_edge_geom(const float* __restrict__ Xn, const int* __restrict__ src,
                            float* __restrict__ diffb, float* __restrict__ normb){
  int e = blockIdx.x * blockDim.x + threadIdx.x;
  if (e >= cE) return;
  int d = e / cKN, s = src[e];
  float dx = Xn[s*3]   - Xn[d*3];
  float dy = Xn[s*3+1] - Xn[d*3+1];
  float dz = Xn[s*3+2] - Xn[d*3+2];
  diffb[e*3] = dx; diffb[e*3+1] = dy; diffb[e*3+2] = dz;
  normb[e] = sqrtf(dx*dx + dy*dy + dz*dz) + 1e-8f;
}

__global__ void k_gather(const bf16* __restrict__ hbf, const int* __restrict__ src,
                         const int* __restrict__ Seg, const float* __restrict__ normb,
                         bf16* __restrict__ MI){
  int idx = blockIdx.x * blockDim.x + threadIdx.x;
  if (idx >= cE * cEINP) return;
  int e = idx / cEINP, col = idx - e * cEINP;
  int dn = e / cKN;
  bf16 v;
  if (col < cH){
    v = hbf[(size_t)src[e]*cH + col];
  } else if (col < 2*cH){
    v = hbf[(size_t)dn*cH + (col - cH)];
  } else if (col < 2*cH + cRBF){
    int k = col - 2*cH;
    float c = (float)k * (1.f / 19.f);
    float t = (normb[e] - c) * 20.f;
    v = (bf16)__expf(-0.5f * t * t);
  } else if (col < cEIN){
    int cls = 2 * Seg[src[e]] + Seg[dn];
    v = (bf16)(((col - 2*cH - cRBF) == cls) ? 1.f : 0.f);
  } else v = (bf16)0.f;
  MI[idx] = v;
}

__global__ void k_reduce9(const bf16* __restrict__ m, bf16* __restrict__ summ){
  int idx = blockIdx.x * blockDim.x + threadIdx.x;
  if (idx >= cN * cH) return;
  int n = idx >> 7, c = idx & 127;
  float s = 0.f;
#pragma unroll
  for (int j = 0; j < cKN; j++) s += (float)m[(size_t)(n*cKN + j)*cH + c];
  summ[idx] = (bf16)s;
}

__global__ void k_v_update(const float* __restrict__ vold, float* __restrict__ vnew,
                           const bf16* __restrict__ g, const float* __restrict__ diffb,
                           const int* __restrict__ src){
  int idx = blockIdx.x * blockDim.x + threadIdx.x;
  if (idx >= cN * cH) return;
  int n = idx >> 7, hh = idx & 127;
  float a0 = vold[(size_t)idx*3], a1 = vold[(size_t)idx*3+1], a2 = vold[(size_t)idx*3+2];
#pragma unroll
  for (int j = 0; j < cKN; j++){
    int e = n*cKN + j, s = src[e];
    float g1 = (float)g[(size_t)e*cH2 + hh];
    float g2 = (float)g[(size_t)e*cH2 + cH + hh];
    size_t vs = ((size_t)s*cH + hh) * 3;
    a0 += g1*diffb[e*3]   + g2*vold[vs];
    a1 += g1*diffb[e*3+1] + g2*vold[vs+1];
    a2 += g1*diffb[e*3+2] + g2*vold[vs+2];
  }
  vnew[(size_t)idx*3] = a0; vnew[(size_t)idx*3+1] = a1; vnew[(size_t)idx*3+2] = a2;
}

__global__ __launch_bounds__(128)
void k_seg_mean(const float* __restrict__ hf, float* __restrict__ means){
  int bi = blockIdx.x, b = bi >> 1, side = bi & 1, h = threadIdx.x;
  int rb = b*cNN + side*cNR;
  float s = 0.f;
  for (int r = 0; r < cNR; r++) s += hf[(size_t)(rb + r)*cH + h];
  means[(size_t)bi*cH + h] = s / (float)cNR;
}

__global__ void k_att_q(const float* __restrict__ Watt_i, const float* __restrict__ means,
                        float* __restrict__ qv){
  int idx = blockIdx.x * blockDim.x + threadIdx.x;
  if (idx >= cBS * 2 * cH) return;
  int b = idx >> 8, rem = idx & 255, side = rem >> 7, h = rem & 127;
  const float* mv = means + (size_t)(b*2 + (1 - side))*cH;
  float s = 0.f;
  for (int k = 0; k < cH; k++) s += Watt_i[(size_t)h*cH + k] * mv[k];
  qv[idx] = s;
}

__global__ __launch_bounds__(128)
void k_gate(float* __restrict__ hf, bf16* __restrict__ hbf,
            const float* __restrict__ actf, const float* __restrict__ qv){
  __shared__ float red[128];
  int n = blockIdx.x, t = threadIdx.x;
  int b = n >> 12, side = (n >> 11) & 1;
  float hv = hf[(size_t)n*cH + t];
  red[t] = hv * qv[(size_t)(b*2 + side)*cH + t];
  __syncthreads();
  for (int s = 64; s > 0; s >>= 1){ if (t < s) red[t] += red[t + s]; __syncthreads(); }
  float g = sigmf(red[0]);
  float nh = hv + g * actf[(size_t)n*cH + t];
  hf[(size_t)n*cH + t] = nh;
  hbf[(size_t)n*cH + t] = (bf16)nh;
}

__global__ void k_split_v(const float* __restrict__ v, bf16* __restrict__ p0,
                          bf16* __restrict__ p1, bf16* __restrict__ p2){
  int idx = blockIdx.x * blockDim.x + threadIdx.x;
  if (idx >= cN * cH) return;
  p0[idx] = (bf16)v[(size_t)idx*3];
  p1[idx] = (bf16)v[(size_t)idx*3+1];
  p2[idx] = (bf16)v[(size_t)idx*3+2];
}

__global__ void k_build_cat(const bf16* __restrict__ hbf, const float* __restrict__ v10,
                            const float* __restrict__ v11, const float* __restrict__ v12,
                            bf16* __restrict__ cat){
  int idx = blockIdx.x * blockDim.x + threadIdx.x;
  if (idx >= cN * cH2) return;
  int n = idx >> 8, c = idx & 255;
  if (c < cH){
    cat[idx] = hbf[(size_t)n*cH + c];
  } else {
    size_t o = (size_t)n*cH + (c - cH);
    float s = v10[o]*v10[o] + v11[o]*v11[o] + v12[o]*v12[o] + 1e-8f;
    cat[idx] = (bf16)sqrtf(s);
  }
}

__global__ void k_split_sg(const float* __restrict__ sg, const float* __restrict__ v20,
                           const float* __restrict__ v21, const float* __restrict__ v22,
                           float* __restrict__ hfo, float* __restrict__ vf0,
                           float* __restrict__ vf1, float* __restrict__ vf2){
  int idx = blockIdx.x * blockDim.x + threadIdx.x;
  if (idx >= cN * cH) return;
  int n = idx >> 7, k = idx & 127;
  hfo[idx] = sg[(size_t)n*cH2 + k];
  float g = sg[(size_t)n*cH2 + cH + k];
  vf0[idx] = g * v20[idx];
  vf1[idx] = g * v21[idx];
  vf2[idx] = g * v22[idx];
}

// ---------------- per-batch geometry head ----------------
__device__ __forceinline__ float blkReduce(float v, float* red, int tid){
  red[tid] = v; __syncthreads();
  for (int s = 128; s > 0; s >>= 1){ if (tid < s) red[tid] += red[tid + s]; __syncthreads(); }
  float r = red[0]; __syncthreads();
  return r;
}

__global__ __launch_bounds__(256)
void k_head(const float* __restrict__ hf, const float* __restrict__ vf0,
            const float* __restrict__ vf1, const float* __restrict__ vf2,
            const float* __restrict__ Xn, const float* __restrict__ oXn,
            const float* __restrict__ kpn, const float* __restrict__ tkpn,
            const float* __restrict__ rot, const float* __restrict__ trans,
            const float* __restrict__ meanp, const float* __restrict__ stdp,
            const float* __restrict__ Wf,
            const float* __restrict__ Wr1, const float* __restrict__ br1,
            const float* __restrict__ Wr2, const float* __restrict__ br2,
            const float* __restrict__ Wl1, const float* __restrict__ bl1,
            const float* __restrict__ Wl2, const float* __restrict__ bl2,
            const float* __restrict__ Weq_re, const float* __restrict__ Weq_li,
            float* __restrict__ accum)
{
  const int b = blockIdx.x, tid = threadIdx.x;
  __shared__ float red[256];
  __shared__ float mr[cH], ml[cH], t1s[cH], t2s[cH];
  __shared__ float xrow[cH], yrow[cH];
  __shared__ float S[3*cH];
  __shared__ float Yb[30];
  __shared__ float P1[cKP*3], P2[cKP*3];
  __shared__ float Rm[2][9], Tm[2][3], csh[2][3];
  __shared__ float inv1[4], inv2[4];
  __shared__ float Lam[3], reprim[3], liprim[3], Rr3[9], Rf[9], Tf[3];
  __shared__ float gsh, kfitsh, docksh;

  const float sm  = (stdp[0] + stdp[1] + stdp[2]) * (1.f / 3.f);
  const float thr = 1.f / sm;
  const int rb0 = b * cNN;
  const int rb1 = b * cNN + cNR;

  // segment means of h and coordinates
  { int side = tid >> 7, h = tid & 127;
    int rb = side ? rb1 : rb0;
    float s = 0.f;
    for (int r = 0; r < cNR; r++) s += hf[(size_t)(rb + r)*cH + h];
    (side ? ml : mr)[h] = s / (float)cNR;
  }
  if (tid < 6){
    int side = tid / 3, d = tid - side*3;
    int rb = side ? rb1 : rb0;
    float s = 0.f;
    for (int r = 0; r < cNR; r++) s += Xn[(size_t)(rb + r)*3 + d];
    csh[side][d] = s / (float)cNR;
  }
  __syncthreads();

  // t1 = Wf @ mean(hl), t2 = Wf @ mean(hr)
  { int h = tid & 127, side = tid >> 7;
    const float* mv = side ? mr : ml;
    float s = 0.f;
    for (int k = 0; k < cH; k++) s += Wf[(size_t)h*cH + k] * mv[k];
    (side ? t2s : t1s)[h] = s;
  }
  if (tid < 4){ inv1[tid] = 0.f; inv2[tid] = 0.f; }
  __syncthreads();

  // invariant heads
  for (int side = 0; side < 2; side++){
    const int rb = side ? rb1 : rb0;
    const float* tg = side ? t2s : t1s;
    const float* W1 = side ? Wl1 : Wr1;  const float* b1 = side ? bl1 : br1;
    const float* W2 = side ? Wl2 : Wr2;  const float* b2 = side ? bl2 : br2;
    for (int r = 0; r < cNR; r++){
      const float* hrow = hf + (size_t)(rb + r)*cH;
      float pv = (tid < cH) ? hrow[tid] * tg[tid] : 0.f;
      float dot = blkReduce(pv, red, tid);
      if (tid == 0) gsh = sigmf(dot);
      __syncthreads();
      if (tid < cH) xrow[tid] = hrow[tid] * gsh;
      __syncthreads();
      if (tid < cH){
        float a = b1[tid];
        for (int h = 0; h < cH; h++) a += xrow[h] * W1[(size_t)h*cH + tid];
        yrow[tid] = siluf(a);
      }
      __syncthreads();
      if (tid < 4){
        float z = b2[tid];
        for (int k = 0; k < cH; k++) z += yrow[k] * W2[k*4 + tid];
        if (side) inv2[tid] += z; else inv1[tid] += z;
      }
      __syncthreads();
    }
  }

  // equivariant heads: S[h][c] = sum_n v[n,h,c]*h[n,h]; Y = Weq^T S / SCALE; MGS
  for (int side = 0; side < 2; side++){
    const int rb = side ? rb1 : rb0;
    for (int idx = tid; idx < 3*cH; idx += 256){
      int c = idx >> 7, h = idx & 127;
      const float* vp = (c == 0) ? vf0 : (c == 1) ? vf1 : vf2;
      float s = 0.f;
      for (int r = 0; r < cNR; r++){
        size_t o = (size_t)(rb + r)*cH + h;
        s += vp[o] * hf[o];
      }
      S[idx] = s;
    }
    __syncthreads();
    if (tid < 30){
      int c = tid / 10, k = tid - c*10;
      const float* Wq = side ? Weq_li : Weq_re;
      float s = 0.f;
      for (int h = 0; h < cH; h++) s += Wq[h*10 + k] * S[c*cH + h];
      Yb[k*3 + c] = s / cSCALE;
    }
    __syncthreads();
    if (tid == 0){
      float P[9], U[9], tv[3];
      for (int i = 0; i < 3; i++)
        for (int c = 0; c < 3; c++)
          P[i*3+c] = Yb[i*3+c] + Yb[(i+3)*3+c] + Yb[(i+6)*3+c];
      for (int c = 0; c < 3; c++) tv[c] = Yb[27 + c];
      const float eps = 1e-8f;
      float n0 = sqrtf(P[0]*P[0] + P[1]*P[1] + P[2]*P[2]) + eps;
      for (int c = 0; c < 3; c++) U[c] = P[c] / n0;
      float d10 = P[3]*U[0] + P[4]*U[1] + P[5]*U[2];
      float w1[3];
      for (int c = 0; c < 3; c++) w1[c] = P[3+c] - d10*U[c];
      float n1 = sqrtf(w1[0]*w1[0] + w1[1]*w1[1] + w1[2]*w1[2]) + eps;
      for (int c = 0; c < 3; c++) U[3+c] = w1[c] / n1;
      float d20 = P[6]*U[0] + P[7]*U[1] + P[8]*U[2];
      float d21 = P[6]*U[3] + P[7]*U[4] + P[8]*U[5];
      float w2[3];
      for (int c = 0; c < 3; c++) w2[c] = P[6+c] - d20*U[c] - d21*U[3+c];
      float n2 = sqrtf(w2[0]*w2[0] + w2[1]*w2[1] + w2[2]*w2[2]) + eps;
      for (int c = 0; c < 3; c++) U[6+c] = w2[c] / n2;
      for (int k = 0; k < 9; k++) Rm[side][k] = U[k];
      for (int k = 0; k < 3; k++){
        float s = 0.f;
        for (int d = 0; d < 3; d++) s += (tv[d] + csh[side][d]) * U[d*3 + k];
        Tm[side][k] = -s;
      }
    }
    __syncthreads();
  }

  if (tid == 0){
    float e0 = fabsf(inv1[0] + inv2[0]);
    float e1 = fabsf(inv1[1] + inv2[1]);
    float e2 = fabsf(inv1[2] + inv2[2]);
    Lam[0] = e0; Lam[1] = e1; Lam[2] = 0.f;
    reprim[0] = 0.f; reprim[1] = 0.f; reprim[2] = -e2;
    liprim[0] = 0.f; liprim[1] = 0.f; liprim[2] =  e2;
    float th = inv1[3] - inv2[3];
    float c_ = cosf(th), s_ = sinf(th);
    Rr3[0] = c_;  Rr3[1] = s_;  Rr3[2] = 0.f;
    Rr3[3] = -s_; Rr3[4] = c_;  Rr3[5] = 0.f;
    Rr3[6] = 0.f; Rr3[7] = 0.f; Rr3[8] = 1.f;
  }
  __syncthreads();

  // keypoint transforms
  if (tid < cKP){
    const float* tk = tkpn + (size_t)(b*cKP + tid)*3;
    const float* kk = kpn  + (size_t)(b*cKP + tid)*3;
    for (int k = 0; k < 3; k++){
      float s1 = Tm[0][k], s2 = Tm[1][k];
      for (int d = 0; d < 3; d++){
        s1 += tk[d] * Rm[0][d*3 + k];
        s2 += kk[d] * Rm[1][d*3 + k];
      }
      P1[tid*3 + k] = s1; P2[tid*3 + k] = s2;
    }
  }
  __syncthreads();

  float hu1 = 0.f, hu2 = 0.f, o1l = 0.f, o1c = 0.f, o2l = 0.f, o2c = 0.f;
  if (tid < cKP){
    float kf1 = 0.f, kf2 = 0.f;
    for (int k = 0; k < 3; k++){
      float a = P1[tid*3+k], c = P2[tid*3+k];
      kf1 += a*a*Lam[k] + a*reprim[k];
      kf2 += c*c*Lam[k] + c*liprim[k];
    }
    hu1 = huberf(kf1); hu2 = huberf(kf2);
    float z1 = P1[tid*3+2];
    if (z1 < -thr){ float d = z1 + thr; o1l = d*d; o1c = 1.f; }
    else if (z1 > thr){ float d = z1 - thr; o1l = d*d; o1c = 1.f; }
    float z2 = P2[tid*3+2];
    if (z2 < -thr){ float d = z2 + thr; o2l = d*d; o2c = 1.f; }
    else if (z2 > thr){ float d = z2 - thr; o2l = d*d; o2c = 1.f; }
  }
  float fit_acc = 0.f;
  fit_acc += blkReduce(hu1, red, tid) / (float)cKP;
  fit_acc += blkReduce(hu2, red, tid) / (float)cKP;
  { float l = blkReduce(o1l, red, tid); float c = blkReduce(o1c, red, tid);
    fit_acc += (c > 0.f) ? l / fmaxf(c, 1.f) : 0.f; }
  { float l = blkReduce(o2l, red, tid); float c = blkReduce(o2c, red, tid);
    fit_acc += (c > 0.f) ? l / fmaxf(c, 1.f) : 0.f; }

  // clash terms over nodes
  float hc1 = 0.f, hc2 = 0.f;
  for (int r = tid; r < cNR; r += 256){
    const float* xr = Xn + (size_t)(rb0 + r)*3;
    const float* xl = Xn + (size_t)(rb1 + r)*3;
    float cf1 = 0.f, cf2 = 0.f;
    for (int k = 0; k < 3; k++){
      float x1 = Tm[0][k] + xr[0]*Rm[0][k] + xr[1]*Rm[0][3+k] + xr[2]*Rm[0][6+k];
      float x2 = Tm[1][k] + xl[0]*Rm[1][k] + xl[1]*Rm[1][3+k] + xl[2]*Rm[1][6+k];
      cf1 += x1*x1*Lam[k] + x1*liprim[k];
      cf2 += x2*x2*Lam[k] + x2*liprim[k];
    }
    float a1 = fmaxf(-cf1, 0.f);
    hc1 += (a1 < 1.f) ? 0.5f*a1*a1 : a1 - 0.5f;
    float a2 = fmaxf(cf2, 0.f);
    hc2 += (a2 < 1.f) ? 0.5f*a2*a2 : a2 - 0.5f;
  }
  fit_acc += blkReduce(hc1, red, tid) / (float)cNR;
  fit_acc += blkReduce(hc2, red, tid) / (float)cNR;

  // 2D Kabsch between P1[:,:2] and P2[:,:2]
  float px = (tid < cKP) ? P1[tid*3]   : 0.f;
  float py = (tid < cKP) ? P1[tid*3+1] : 0.f;
  float qx = (tid < cKP) ? P2[tid*3]   : 0.f;
  float qy = (tid < cKP) ? P2[tid*3+1] : 0.f;
  float mpx = blkReduce(px, red, tid) / (float)cKP;
  float mpy = blkReduce(py, red, tid) / (float)cKP;
  float mqx = blkReduce(qx, red, tid) / (float)cKP;
  float mqy = blkReduce(qy, red, tid) / (float)cKP;
  float c00 = 0.f, c01 = 0.f, c10 = 0.f, c11 = 0.f;
  if (tid < cKP){
    float p0 = px - mpx, p1v = py - mpy, q0 = qx - mqx, q1 = qy - mqy;
    c00 = p0*q0; c01 = p0*q1; c10 = p1v*q0; c11 = p1v*q1;
  }
  float M00 = blkReduce(c00, red, tid);
  float M01 = blkReduce(c01, red, tid);
  float M10 = blkReduce(c10, red, tid);
  float M11 = blkReduce(c11, red, tid);
  if (tid == 0){
    float a = M00 + M11, bq = M10 - M01;
    float rr = sqrtf(a*a + bq*bq) + 1e-12f;
    float ck = a/rr, sk = bq/rr;
    float d0 = Rr3[0] - ck, d1 = Rr3[1] + sk, d2 = Rr3[3] - sk, d3 = Rr3[4] - ck;
    kfitsh = (d0*d0 + d1*d1 + d2*d2 + d3*d3) * 0.25f;
    // combined transform
    for (int i = 0; i < 3; i++)
      for (int j = 0; j < 3; j++){
        float s = 0.f;
        for (int p = 0; p < 3; p++)
          for (int q = 0; q < 3; q++)
            s += Rm[0][i*3+p] * Rr3[p*3+q] * Rm[1][j*3+q];
        Rf[i*3+j] = s;
      }
    float u[3];
    for (int q = 0; q < 3; q++){
      float s = 0.f;
      for (int p = 0; p < 3; p++) s += Tm[0][p] * Rr3[p*3+q];
      u[q] = s - Tm[1][q];
    }
    for (int j = 0; j < 3; j++){
      float s = 0.f;
      for (int q = 0; q < 3; q++) s += u[q] * Rm[1][j*3+q];
      Tf[j] = s;
    }
    float sR = 0.f;
    for (int i = 0; i < 3; i++)
      for (int j = 0; j < 3; j++){
        float d = Rf[i*3+j] - rot[b*9 + j*3 + i];
        sR += d*d;
      }
    float dk = sR / 9.f;
    float ct[3];
    for (int d = 0; d < 3; d++) ct[d] = meanp[d] / sm;
    float sT = 0.f;
    for (int j = 0; j < 3; j++){
      float tg = 0.f;
      for (int d = 0; d < 3; d++) tg += (ct[d] - trans[b*3+d]) * rot[b*9 + j*3 + d];
      tg -= ct[j];
      float dv = Tf[j] - tg;
      sT += dv*dv;
    }
    docksh = dk + sT / 3.f;
  }
  __syncthreads();

  // rmsd
  float rs = 0.f;
  for (int r = tid; r < cNR; r += 256){
    const float* xr = Xn + (size_t)(rb0 + r)*3;
    const float* ox = oXn + (size_t)(rb0 + r)*3;
    for (int k = 0; k < 3; k++){
      float y = Tf[k] + xr[0]*Rf[k] + xr[1]*Rf[3+k] + xr[2]*Rf[6+k];
      float d = y - ox[k];
      rs += d*d;
    }
  }
  float rmsd = blkReduce(rs, red, tid) / (float)(cNR * 3);

  if (tid == 0){
    atomicAdd(accum + 0, (fit_acc + kfitsh) / (float)cBS);
    atomicAdd(accum + 1, docksh / (float)cBS);
    atomicAdd(accum + 2, rmsd / (float)cBS);
  }
}

__global__ void k_finalize(const float* __restrict__ accum, float* __restrict__ out){
  if (blockIdx.x == 0 && threadIdx.x == 0){
    float fit = accum[0], dock = accum[1], rmsd = accum[2];
    out[0] = 0.5f*fit + dock + rmsd;
    out[1] = fit; out[2] = dock; out[3] = 0.f; out[4] = rmsd;
  }
}

} // anonymous namespace

extern "C" void kernel_launch(void* const* d_in, const int* in_sizes, int n_in,
                              void* d_out, int out_size, void* d_ws, size_t ws_size,
                              hipStream_t stream) {
  (void)in_sizes; (void)n_in; (void)out_size; (void)ws_size;
  const float* X         = (const float*)d_in[0];
  const float* node_attr = (const float*)d_in[1];
  const int*   Seg       = (const int*)d_in[2];
  const int*   bid       = (const int*)d_in[3];
  const float* center    = (const float*)d_in[4];
  const float* keypoints = (const float*)d_in[5];
  const int*   k_bid     = (const int*)d_in[6];
  const float* rot       = (const float*)d_in[7];
  const float* trans     = (const float*)d_in[8];
  const float* in_W1 = (const float*)d_in[9];   const float* in_b1 = (const float*)d_in[10];
  const float* in_W2 = (const float*)d_in[11];  const float* in_b2 = (const float*)d_in[12];
  const float* We1   = (const float*)d_in[13];  const float* be1   = (const float*)d_in[14];
  const float* We2   = (const float*)d_in[15];  const float* be2   = (const float*)d_in[16];
  const float* Wh    = (const float*)d_in[17];  const float* bh    = (const float*)d_in[18];
  const float* Wv    = (const float*)d_in[19];
  const float* Wa1   = (const float*)d_in[20];  const float* ba1   = (const float*)d_in[21];
  const float* Wa2   = (const float*)d_in[22];  const float* ba2   = (const float*)d_in[23];
  const float* Watt  = (const float*)d_in[24];
  const float* Wv1   = (const float*)d_in[25];  const float* Wv2   = (const float*)d_in[26];
  const float* Wg1   = (const float*)d_in[27];  const float* bg1   = (const float*)d_in[28];
  const float* Wg2   = (const float*)d_in[29];  const float* bg2   = (const float*)d_in[30];
  const float* Wr1   = (const float*)d_in[31];  const float* br1   = (const float*)d_in[32];
  const float* Wr2   = (const float*)d_in[33];  const float* br2   = (const float*)d_in[34];
  const float* Wl1   = (const float*)d_in[35];  const float* bl1   = (const float*)d_in[36];
  const float* Wl2   = (const float*)d_in[37];  const float* bl2   = (const float*)d_in[38];
  const float* Wf    = (const float*)d_in[39];
  const float* Weq_re = (const float*)d_in[40]; const float* Weq_li = (const float*)d_in[41];
  const float* meanp = (const float*)d_in[42];  const float* stdp  = (const float*)d_in[43];
  float* out = (float*)d_out;

  // ---- workspace layout ----
  char* basep = (char*)d_ws;
  size_t off = 0;
  auto alloc = [&](size_t bytes) -> char* {
    off = (off + 255) & ~(size_t)255;
    char* p = basep + off;
    off += bytes;
    return p;
  };
  bf16* w_in1T = (bf16*)alloc((size_t)cH*32*2);
  bf16* w_in2T = (bf16*)alloc((size_t)cH*cH*2);
  bf16* w_e1T  = (bf16*)alloc((size_t)cL*cH*cEINP*2);
  bf16* w_e2T  = (bf16*)alloc((size_t)cL*cH*cH*2);
  bf16* w_hT   = (bf16*)alloc((size_t)cL*cH*cH*2);
  bf16* w_vT   = (bf16*)alloc((size_t)cL*cH2*cH*2);
  bf16* w_a1T  = (bf16*)alloc((size_t)cL*cH*cH*2);
  bf16* w_a2T  = (bf16*)alloc((size_t)cL*cH*cH*2);
  bf16* w_v1T  = (bf16*)alloc((size_t)cH*cH*2);
  bf16* w_v2T  = (bf16*)alloc((size_t)cH*cH*2);
  bf16* w_g1T  = (bf16*)alloc((size_t)cH2*cH2*2);
  bf16* w_g2T  = (bf16*)alloc((size_t)cH2*cH2*2);
  float* Xn    = (float*)alloc((size_t)cN*3*4);
  float* oXn   = (float*)alloc((size_t)cN*3*4);
  float* kpn   = (float*)alloc((size_t)cBS*cKP*3*4);
  float* tkpn  = (float*)alloc((size_t)cBS*cKP*3*4);
  int*   srcb  = (int*)alloc((size_t)cE*4);
  float* diffb = (float*)alloc((size_t)cE*3*4);
  float* normb = (float*)alloc((size_t)cE*4);
  bf16*  hbf   = (bf16*)alloc((size_t)cN*cH*2);
  float* hf    = (float*)alloc((size_t)cN*cH*4);
  bf16*  A0    = (bf16*)alloc((size_t)cN*32*2);
  bf16*  nb1   = (bf16*)alloc((size_t)cN*cH*2);      // embed intermediate / summed messages
  char*  MIreg = alloc((size_t)cE*cEINP*2);          // msg_in; reused: g, head planes
  bf16*  M1    = (bf16*)alloc((size_t)cE*cH*2);      // reused: cat/sgA/sg
  bf16*  mbuf  = (bf16*)alloc((size_t)cE*cH*2);      // reused: vfin planes
  float* vA    = (float*)alloc((size_t)cN*cH*3*4);
  float* vB    = (float*)alloc((size_t)cN*cH*3*4);
  float* meansb= (float*)alloc((size_t)cBS*2*cH*4);
  float* qvb   = (float*)alloc((size_t)cBS*2*cH*4);
  bf16*  actb  = (bf16*)alloc((size_t)cN*cH*2);
  float* actf  = (float*)alloc((size_t)cN*cH*4);
  float* accum = (float*)alloc(3*4);

  bf16*  MI    = (bf16*)MIreg;
  bf16*  gbuf  = (bf16*)MIreg;                        // E x 256 bf16 (after MI is dead)
  // head overlays
  bf16*  p0 = (bf16*)MIreg;
  bf16*  p1 = p0 + (size_t)cN*cH;
  bf16*  p2 = p1 + (size_t)cN*cH;
  float* v1p = (float*)(MIreg + (size_t)3*cN*cH*2);
  float* v2p = v1p + (size_t)3*cN*cH;
  bf16*  cat = (bf16*)M1;
  bf16*  sgA = cat + (size_t)cN*cH2;
  float* sg  = (float*)((char*)M1 + (size_t)2*cN*cH2*2);
  float* vfin = (float*)mbuf;

  auto g1 = [](int n){ return (n + 255) / 256; };

  // init
  k_zero<<<g1(cN*cH*3), 256, 0, stream>>>(vA, cN*cH*3);
  k_zero<<<1, 32, 0, stream>>>(accum, 3);

  // weight conversion to bf16, transposed [Nout x Kpad]
  auto cvtT = [&](const float* W, bf16* WT, int K, int Nn, int Kp){
    k_wT<<<g1(Nn*Kp), 256, 0, stream>>>(W, WT, K, Nn, Kp);
  };
  cvtT(in_W1, w_in1T, 32, cH, 32);
  cvtT(in_W2, w_in2T, cH, cH, cH);
  for (int i = 0; i < cL; i++){
    cvtT(We1 + (size_t)i*cEIN*cH, w_e1T + (size_t)i*cH*cEINP, cEIN, cH, cEINP);
    cvtT(We2 + (size_t)i*cH*cH,   w_e2T + (size_t)i*cH*cH,   cH, cH, cH);
    cvtT(Wh  + (size_t)i*cH*cH,   w_hT  + (size_t)i*cH*cH,   cH, cH, cH);
    cvtT(Wv  + (size_t)i*cH*cH2,  w_vT  + (size_t)i*cH2*cH,  cH, cH2, cH);
    cvtT(Wa1 + (size_t)i*cH*cH,   w_a1T + (size_t)i*cH*cH,   cH, cH, cH);
    cvtT(Wa2 + (size_t)i*cH*cH,   w_a2T + (size_t)i*cH*cH,   cH, cH, cH);
  }
  cvtT(Wv1, w_v1T, cH, cH, cH);
  cvtT(Wv2, w_v2T, cH, cH, cH);
  cvtT(Wg1, w_g1T, cH2, cH2, cH2);
  cvtT(Wg2, w_g2T, cH2, cH2, cH2);

  // coordinates / graph
  k_coords_nodes<<<g1(cN), 256, 0, stream>>>(X, Seg, bid, center, rot, trans, meanp, stdp, Xn, oXn);
  k_coords_kp<<<g1(cBS*cKP), 256, 0, stream>>>(keypoints, k_bid, center, rot, trans, meanp, stdp, kpn, tkpn);
  k_knn<<<g1(cN), 256, 0, stream>>>(Xn, srcb);
  k_edge_geom<<<g1(cE), 256, 0, stream>>>(Xn, srcb, diffb, normb);

  // input embedding
  k_cvt_bf16<<<g1(cN*32), 256, 0, stream>>>(node_attr, A0, cN*32);
  k_gemm<EP_BIAS|EP_SILU|EP_OUTB><<<(cN/128)*(cH/128), 128, 0, stream>>>(
      A0, w_in1T, in_b1, nullptr, nullptr, nb1, cN, cH, 32);
  k_gemm<EP_BIAS|EP_OUTF|EP_OUTB><<<(cN/128)*(cH/128), 128, 0, stream>>>(
      nb1, w_in2T, in_b2, nullptr, hf, hbf, cN, cH, cH);

  float* vold = vA;
  float* vnew = vB;
  for (int i = 0; i < cL; i++){
    k_gather<<<g1(cE*cEINP), 256, 0, stream>>>(hbf, srcb, Seg, normb, MI);
    k_gemm<EP_BIAS|EP_SILU|EP_OUTB><<<(cE/128)*(cH/128), 128, 0, stream>>>(
        MI, w_e1T + (size_t)i*cH*cEINP, be1 + (size_t)i*cH, nullptr, nullptr, M1, cE, cH, cEINP);
    k_gemm<EP_BIAS|EP_SILU|EP_OUTB><<<(cE/128)*(cH/128), 128, 0, stream>>>(
        M1, w_e2T + (size_t)i*cH*cH, be2 + (size_t)i*cH, nullptr, nullptr, mbuf, cE, cH, cH);
    k_gemm<EP_OUTB><<<(cE/128)*(cH2/128), 128, 0, stream>>>(
        mbuf, w_vT + (size_t)i*cH2*cH, nullptr, nullptr, nullptr, gbuf, cE, cH2, cH);
    k_reduce9<<<g1(cN*cH), 256, 0, stream>>>(mbuf, nb1);
    k_gemm<EP_BIAS|EP_RES|EP_OUTF|EP_OUTB><<<(cN/128)*(cH/128), 128, 0, stream>>>(
        nb1, w_hT + (size_t)i*cH*cH, bh + (size_t)i*cH, hf, hf, hbf, cN, cH, cH);
    k_v_update<<<g1(cN*cH), 256, 0, stream>>>(vold, vnew, gbuf, diffb, srcb);
    { float* t = vold; vold = vnew; vnew = t; }
    k_seg_mean<<<cBS*2, 128, 0, stream>>>(hf, meansb);
    k_att_q<<<g1(cBS*2*cH), 256, 0, stream>>>(Watt + (size_t)i*cH*cH, meansb, qvb);
    k_gemm<EP_BIAS|EP_SILU|EP_OUTB><<<(cN/128)*(cH/128), 128, 0, stream>>>(
        hbf, w_a1T + (size_t)i*cH*cH, ba1 + (size_t)i*cH, nullptr, nullptr, actb, cN, cH, cH);
    k_gemm<EP_BIAS|EP_OUTF><<<(cN/128)*(cH/128), 128, 0, stream>>>(
        actb, w_a2T + (size_t)i*cH*cH, ba2 + (size_t)i*cH, nullptr, actf, nullptr, cN, cH, cH);
    k_gate<<<cN, 128, 0, stream>>>(hf, hbf, actf, qvb);
  }

  // head
  k_split_v<<<g1(cN*cH), 256, 0, stream>>>(vold, p0, p1, p2);
  for (int c = 0; c < 3; c++){
    bf16* pc = (c == 0) ? p0 : (c == 1) ? p1 : p2;
    k_gemm<EP_OUTF><<<(cN/128)*(cH/128), 128, 0, stream>>>(
        pc, w_v1T, nullptr, nullptr, v1p + (size_t)c*cN*cH, nullptr, cN, cH, cH);
    k_gemm<EP_OUTF><<<(cN/128)*(cH/128), 128, 0, stream>>>(
        pc, w_v2T, nullptr, nullptr, v2p + (size_t)c*cN*cH, nullptr, cN, cH, cH);
  }
  k_build_cat<<<g1(cN*cH2), 256, 0, stream>>>(
      hbf, v1p, v1p + (size_t)cN*cH, v1p + (size_t)2*cN*cH, cat);
  k_gemm<EP_BIAS|EP_SILU|EP_OUTB><<<(cN/128)*(cH2/128), 128, 0, stream>>>(
      cat, w_g1T, bg1, nullptr, nullptr, sgA, cN, cH2, cH2);
  k_gemm<EP_BIAS|EP_OUTF><<<(cN/128)*(cH2/128), 128, 0, stream>>>(
      sgA, w_g2T, bg2, nullptr, sg, nullptr, cN, cH2, cH2);
  k_split_sg<<<g1(cN*cH), 256, 0, stream>>>(
      sg, v2p, v2p + (size_t)cN*cH, v2p + (size_t)2*cN*cH,
      hf, vfin, vfin + (size_t)cN*cH, vfin + (size_t)2*cN*cH);

  k_head<<<cBS, 256, 0, stream>>>(
      hf, vfin, vfin + (size_t)cN*cH, vfin + (size_t)2*cN*cH,
      Xn, oXn, kpn, tkpn, rot, trans, meanp, stdp,
      Wf, Wr1, br1, Wr2, br2, Wl1, bl1, Wl2, bl2, Weq_re, Weq_li, accum);
  k_finalize<<<1, 1, 0, stream>>>(accum, out);
}
